// AttnConv2d_56401510531573
// MI455X (gfx1250) — matrix operand order repaired and checked by reading, not compiled, not run on hardware
//
#include <hip/hip_runtime.h>
#include <hip/hip_bf16.h>
#include <stdint.h>

// ---------------- problem constants ----------------
#define B_   8
#define CI   128
#define CO   128
#define H_   192
#define W_   192
#define HW   (H_ * W_)          // 36864
#define KK   1152               // 9 * 128, k-order = r*128 + ci (r = kh*3+kw)
#define KSTEP 32
#define NPIX 64
#define NORM_SCALE_F 0.1816f
#define EPS_F 1e-5f

typedef __attribute__((ext_vector_type(16))) __bf16 v16bf;
typedef __attribute__((ext_vector_type(8)))  float  v8f;

union Frag { v16bf v; uint32_t u[8]; };

__device__ __forceinline__ unsigned short f2bf(float f) {
  uint32_t u = __builtin_bit_cast(uint32_t, f);
  u += 0x7FFFu + ((u >> 16) & 1u);           // round-to-nearest-even
  return (unsigned short)(u >> 16);
}

// CDNA5 async global->LDS copy (ASYNCcnt-tracked), GV addressing.
__device__ __forceinline__ void async_ld_b128(const void* g, void* l) {
  unsigned long long ga = (unsigned long long)(uintptr_t)g;
  unsigned int la = (unsigned int)(uintptr_t)l;   // low 32 bits = LDS offset
  asm volatile("global_load_async_to_lds_b128 %0, %1, off"
               :: "v"(la), "v"(ga) : "memory");
}
__device__ __forceinline__ void wait_async0() {
#if __has_builtin(__builtin_amdgcn_s_wait_asynccnt)
  __builtin_amdgcn_s_wait_asynccnt(0);
#else
  asm volatile("s_wait_asynccnt 0x0" ::: "memory");
#endif
}

// ---------------------------------------------------------------------------
// one-time bf16 conversions
// ---------------------------------------------------------------------------
__global__ __launch_bounds__(256) void cvt_x_bf16(
    const float* __restrict__ src, unsigned short* __restrict__ dst) {
  int i = blockIdx.x * 256 + threadIdx.x;
  dst[i] = f2bf(src[i]);
}
// permute weights [co][ci][r] -> bf16 [co][r*128+ci] (r-major k-order)
__global__ __launch_bounds__(256) void cvt_w_bf16(
    const float* __restrict__ w, unsigned short* __restrict__ dst) {
  int i = blockIdx.x * 256 + threadIdx.x;      // over CO*KK
  int co = i / KK, k = i % KK;
  int r = k >> 7, ci = k & 127;
  dst[i] = f2bf(w[(co * CI + ci) * 9 + r]);
}

// ---------------------------------------------------------------------------
// 3x3 SAME conv as implicit GEMM, bf16 WMMA, f32 accum.
// M=Co(128) x N=64 pixels x K=1152 (r-major).  256 thr = 8 waves;
// wave owns 16 Co rows x 4 N-tiles.  Weights async-staged to LDS (b128),
// im2col x staged with scalar row index per chunk.
// F32OUT=0 -> bf16 out (a1/a2/a3);  F32OUT=1 -> f32 out (av).
// ---------------------------------------------------------------------------
template <int F32OUT>
__global__ __launch_bounds__(256) void conv3x3_wmma(
    const unsigned short* __restrict__ xin,   // bf16 [B][CI][H][W]
    const unsigned short* __restrict__ wgt,   // bf16 [(B)][CO][KK]
    int perBatchW, void* __restrict__ outp)
{
  __shared__ __align__(16) unsigned short lds_w[CO * KSTEP];   // [co][kk]
  __shared__ __align__(16) unsigned short lds_x[NPIX * KSTEP]; // [n][kk]

  const int tid  = threadIdx.x;
  const int lane = tid & 31;
  const int b    = blockIdx.x / (HW / NPIX);
  const int pix0 = (blockIdx.x % (HW / NPIX)) * NPIX;
  const int h0   = pix0 / W_;
  const int w0   = pix0 % W_;
  const int co0  = (tid >> 5) * 16;
  const unsigned short* wb = wgt + (perBatchW ? b * CO * KK : 0);

  v8f acc[4];
  #pragma unroll
  for (int t = 0; t < 4; ++t) acc[t] = (v8f){};

  for (int r = 0; r < 9; ++r) {
    const int kh = r / 3, kw = r % 3;         // scalar per chunk
    const int hs = h0 + kh - 1;
    const bool rowok = ((unsigned)hs < (unsigned)H_);
    const unsigned short* xrow = xin + (b * CI * H_ + hs) * W_;

    for (int cb = 0; cb < CI; cb += KSTEP) {
      const int kb = r * 128 + cb;
      // --- weights: 8192 B tile, contiguous per co row -> async b128 x2 ---
      {
        int ob = tid * 32;                    // byte offset in tile
        int co = ob >> 6;                     // 64 B per co row
        int inner = ob & 63;
        const char* g = (const char*)(wb + co * KK + kb) + inner;
        char* l = (char*)lds_w + ob;
        async_ld_b128(g, l);
        async_ld_b128(g + 16, l + 16);
      }
      // --- im2col x tile: 64 n x 32 ci, scalar (kh,kw,hs) ---
      #pragma unroll
      for (int e = 0; e < 8; ++e) {
        int idx = e * 256 + tid;
        int n = idx >> 5, kkx = idx & 31;
        int ws2 = w0 + n + kw - 1;
        unsigned short val = 0;
        if (rowok && (unsigned)ws2 < (unsigned)W_)
          val = xrow[(cb + kkx) * HW + ws2];
        lds_x[idx] = val;
      }
      wait_async0();
      __syncthreads();

      Frag a;
      #pragma unroll
      for (int v = 0; v < 8; ++v) {
        int kA = 2 * v + (v >= 4 ? 8 : 0) + (lane >= 16 ? 8 : 0);
        a.u[v] = *(const uint32_t*)&lds_w[(co0 + (lane & 15)) * KSTEP + kA];
      }
      #pragma unroll
      for (int t = 0; t < 4; ++t) {
        Frag bf;
        #pragma unroll
        for (int v = 0; v < 8; ++v) {
          int kB = 2 * v + (v >= 4 ? 8 : 0) + (lane >= 16 ? 8 : 0);
          bf.u[v] = *(const uint32_t*)&lds_x[(t * 16 + (lane & 15)) * KSTEP + kB];
        }
        acc[t] = __builtin_amdgcn_wmma_f32_16x16x32_bf16(
            false, a.v, false, bf.v, (short)0, acc[t], false, false);
      }
      __syncthreads();
    }
  }

  #pragma unroll
  for (int t = 0; t < 4; ++t) {
    int n = pix0 + t * 16 + (lane & 15);
    #pragma unroll
    for (int v = 0; v < 8; ++v) {
      int m = v + (lane >= 16 ? 8 : 0);
      if constexpr (F32OUT)
        ((float*)outp)[(b * CO + co0 + m) * HW + n] = acc[t][v];
      else
        ((unsigned short*)outp)[(b * CO + co0 + m) * HW + n] = f2bf(acc[t][v]);
    }
  }
}

// ---------------------------------------------------------------------------
// attn logits: per (b, oh, ow): L[ci,co] = sum_{p,q<64} a1[..,3p+oh,3q+ow]*a2[..]
// 128x128 GEMM, K=4096.  Stored r-major: logits[b][co][off*128+ci].
// ---------------------------------------------------------------------------
__global__ __launch_bounds__(256) void attn_gemm_wmma(
    const unsigned short* __restrict__ a1, const unsigned short* __restrict__ a2,
    float* __restrict__ logits)
{
  __shared__ __align__(16) unsigned short lds_a[CI * KSTEP];
  __shared__ __align__(16) unsigned short lds_b[CO * KSTEP];

  const int tid  = threadIdx.x;
  const int lane = tid & 31;
  const int b    = blockIdx.x / 9;
  const int off  = blockIdx.x % 9;
  const int oh   = off / 3, ow = off % 3;
  const int ci0  = (tid >> 5) * 16;

  v8f acc[8];
  #pragma unroll
  for (int t = 0; t < 8; ++t) acc[t] = (v8f){};

  for (int kb = 0; kb < 4096; kb += KSTEP) {
    #pragma unroll
    for (int e = 0; e < 16; ++e) {
      int idx = e * 256 + tid;
      int cc  = idx >> 5, kk = idx & 31;
      int k   = kb + kk;
      int p = k >> 6, q = k & 63;
      int sp = (3 * p + oh) * W_ + 3 * q + ow;
      lds_a[idx] = a1[(b * CI + cc) * HW + sp];
      lds_b[idx] = a2[(b * CO + cc) * HW + sp];
    }
    __syncthreads();

    Frag a;
    #pragma unroll
    for (int v = 0; v < 8; ++v) {
      int kA = 2 * v + (v >= 4 ? 8 : 0) + (lane >= 16 ? 8 : 0);
      a.u[v] = *(const uint32_t*)&lds_a[(ci0 + (lane & 15)) * KSTEP + kA];
    }
    #pragma unroll
    for (int t = 0; t < 8; ++t) {
      Frag bf;
      #pragma unroll
      for (int v = 0; v < 8; ++v) {
        int kB = 2 * v + (v >= 4 ? 8 : 0) + (lane >= 16 ? 8 : 0);
        bf.u[v] = *(const uint32_t*)&lds_b[(t * 16 + (lane & 15)) * KSTEP + kB];
      }
      acc[t] = __builtin_amdgcn_wmma_f32_16x16x32_bf16(
          false, a.v, false, bf.v, (short)0, acc[t], false, false);
    }
    __syncthreads();
  }

  #pragma unroll
  for (int t = 0; t < 8; ++t) {
    int co = t * 16 + (lane & 15);
    #pragma unroll
    for (int v = 0; v < 8; ++v) {
      int ci = ci0 + v + (lane >= 16 ? 8 : 0);
      logits[(b * CO + co) * KK + off * 128 + ci] = acc[t][v];
    }
  }
}

// ---------------------------------------------------------------------------
// Row softmax over 1152 (scale 1/sqrt(1152)) -> bf16 dynamic kernels
// (same r-major k-order as the conv consumes; softmax is order-invariant)
// ---------------------------------------------------------------------------
__global__ __launch_bounds__(256) void softmax_rows(
    const float* __restrict__ logits, unsigned short* __restrict__ kern)
{
  __shared__ float red[256];
  const int tid = threadIdx.x;
  const int row = blockIdx.x;
  const float* src = logits + row * KK;
  const float scale = rsqrtf((float)KK);

  float m = -1e30f;
  for (int j = tid; j < KK; j += 256) m = fmaxf(m, src[j] * scale);
  red[tid] = m; __syncthreads();
  for (int s = 128; s > 0; s >>= 1) {
    if (tid < s) red[tid] = fmaxf(red[tid], red[tid + s]);
    __syncthreads();
  }
  const float rowmax = red[0];
  __syncthreads();

  float sum = 0.0f;
  for (int j = tid; j < KK; j += 256) sum += __expf(src[j] * scale - rowmax);
  red[tid] = sum; __syncthreads();
  for (int s = 128; s > 0; s >>= 1) {
    if (tid < s) red[tid] += red[tid + s];
    __syncthreads();
  }
  const float inv = 1.0f / red[0];

  for (int j = tid; j < KK; j += 256)
    kern[row * KK + j] = f2bf(__expf(src[j] * scale - rowmax) * inv);
}

// ---------------------------------------------------------------------------
// BN stats of feature_map_stack(av) per OUTPUT channel (stack fused):
// out channel c = 4g+2*i2+j2 gathers input channels [4g,4g+4), parity (i2,j2).
// ---------------------------------------------------------------------------
__global__ __launch_bounds__(256) void bn_stats(
    const float* __restrict__ av, float* __restrict__ stats)
{
  __shared__ float sred[256];
  __shared__ float qred[256];
  const int tid = threadIdx.x;
  const int c = blockIdx.x;
  const int g = c >> 2, i2 = (c >> 1) & 1, j2 = c & 1;
  const int NROWS = B_ * 4 * 96;              // 3072 sample rows of 96
  const int NTOT  = NROWS * 96;

  float s = 0.0f, q = 0.0f;
  for (int row = tid; row < NROWS; row += 256) {
    int b  = row / 384;
    int r2 = row % 384;
    int dc = r2 / 96, ph = r2 % 96;
    const float* p = av + ((b * CO + g * 4 + dc) * H_ + 2 * ph + i2) * W_ + j2;
    #pragma unroll 4
    for (int pw = 0; pw < 96; ++pw) {
      float v = p[2 * pw];
      s += v; q += v * v;
    }
  }
  sred[tid] = s; qred[tid] = q; __syncthreads();
  for (int st = 128; st > 0; st >>= 1) {
    if (tid < st) { sred[tid] += sred[tid + st]; qred[tid] += qred[tid + st]; }
    __syncthreads();
  }
  if (tid == 0) {
    float mean = sred[0] / (float)NTOT;
    float var  = qred[0] / (float)NTOT - mean * mean;
    stats[c]       = mean;
    stats[128 + c] = rsqrtf(var + EPS_F);
  }
}

// ---------------------------------------------------------------------------
// out = conv_momentum*x + 0.1816*(stack(av) - mean)*invstd  (stack as gather)
// ---------------------------------------------------------------------------
__global__ __launch_bounds__(256) void finalize(
    const float* __restrict__ x, const float* __restrict__ av,
    const float* __restrict__ stats, const float* __restrict__ cm_ptr,
    float* __restrict__ out)
{
  int idx = blockIdx.x * 256 + threadIdx.x;
  int b   = idx / (CO * HW);
  int r   = idx % (CO * HW);
  int c   = r / HW;
  int pix = r % HW;
  int y   = pix / W_, xx = pix % W_;
  int g   = c >> 2, i2 = (c >> 1) & 1, j2 = c & 1;
  int i1  = y / 96, ph = y % 96;
  int j1  = xx / 96, pw = xx % 96;
  int cin = g * 4 + i1 * 2 + j1;
  float v = av[((b * CO + cin) * H_ + 2 * ph + i2) * W_ + 2 * pw + j2];
  out[idx] = cm_ptr[0] * x[idx] + NORM_SCALE_F * (v - stats[c]) * stats[128 + c];
}

// ---------------------------------------------------------------------------
extern "C" void kernel_launch(void* const* d_in, const int* in_sizes, int n_in,
                              void* d_out, int out_size, void* d_ws, size_t ws_size,
                              hipStream_t stream) {
  const float* x  = (const float*)d_in[0];
  const float* w1 = (const float*)d_in[1];
  const float* w2 = (const float*)d_in[2];
  const float* w3 = (const float*)d_in[3];
  const float* cm = (const float*)d_in[4];
  float* out = (float*)d_out;

  char* ws = (char*)d_ws;
  const size_t bf = (size_t)B_ * CO * HW * 2;            // 75,497,472 B
  unsigned short* xb  = (unsigned short*)(ws);           // dead after convs
  unsigned short* a1b = (unsigned short*)(ws + bf);      // dead after attn
  unsigned short* a2b = (unsigned short*)(ws + 2 * bf);  // dead after attn
  unsigned short* a3b = (unsigned short*)(ws + 3 * bf);
  float* av = (float*)(ws);                              // 2*bf, aliases xb+a1b
  unsigned short* w1b = (unsigned short*)(ws + 4 * bf);
  unsigned short* w2b = (unsigned short*)(ws + 4 * bf + 294912);
  unsigned short* w3b = (unsigned short*)(ws + 4 * bf + 2 * 294912);
  float* logits = (float*)(ws + 4 * bf + 3 * 294912);    // 4,718,592 B
  unsigned short* kern = (unsigned short*)(ws + 4 * bf + 3 * 294912 + 4718592);
  float* stats = (float*)(ws + 4 * bf + 3 * 294912 + 4718592 + 2359296);

  const dim3 blk(256);
  const int convGrid = B_ * (HW / NPIX);                 // 4608
  const int nElem = B_ * CO * HW;                        // 37,748,736

  cvt_x_bf16<<<nElem / 256, blk, 0, stream>>>(x, xb);
  cvt_w_bf16<<<(CO * KK) / 256, blk, 0, stream>>>(w1, w1b);
  cvt_w_bf16<<<(CO * KK) / 256, blk, 0, stream>>>(w2, w2b);
  cvt_w_bf16<<<(CO * KK) / 256, blk, 0, stream>>>(w3, w3b);

  conv3x3_wmma<0><<<convGrid, blk, 0, stream>>>(xb, w1b, 0, a1b);
  conv3x3_wmma<0><<<convGrid, blk, 0, stream>>>(xb, w2b, 0, a2b);
  conv3x3_wmma<0><<<convGrid, blk, 0, stream>>>(xb, w3b, 0, a3b);
  attn_gemm_wmma<<<B_ * 9, blk, 0, stream>>>(a1b, a2b, logits);
  softmax_rows<<<B_ * CO, blk, 0, stream>>>(logits, kern);
  conv3x3_wmma<1><<<convGrid, blk, 0, stream>>>(a3b, kern, 1, (void*)av);
  bn_stats<<<CO, blk, 0, stream>>>(av, stats);
  finalize<<<nElem / 256, blk, 0, stream>>>(x, av, stats, cm, out);
}
